// InstancesMLPComponentsRankPenalty_15848429322661
// MI455X (gfx1250) — matrix operand order repaired and checked by reading, not compiled, not run on hardware
//
#include <hip/hip_runtime.h>
#include <hip/hip_bf16.h>

// ---------------------------------------------------------------------------
// InstancesMLPComponentsRankPenalty — CDNA5 (gfx1250) WMMA implementation, v2
//
// Shapes: BATCH=128, N_INST=8, D_EMBED=256, D_MLP=1024, K=64, M=32.
// All GEMMs via v_wmma_f32_16x16x32_bf16 (bf16 operands, fp32 accumulate).
//
// v2 changes vs v1 (driven by histogram: VALU bloat from manual bf16 cvt,
// scalar b32 loads for fragments):
//  * native (__bf16) casts instead of integer rounding chains
//  * pre kernels emit the [128x32] intermediates as bf16 pre-swizzled into
//    the WMMA A-fragment layout (transpose through LDS), so post kernels
//    load A with one contiguous 32B vector load per lane, no conversion
//  * pre kernels process 4 components per block -> activation tile converted
//    once per 8 WMMAs; layer-1 post uses full-batch tile so B1 is read once
// ---------------------------------------------------------------------------

typedef __attribute__((ext_vector_type(16))) __bf16 v16bf;
typedef __attribute__((ext_vector_type(8)))  float  v8f;

#define NB   128   // batch
#define NI   8     // instances
#define ND   256   // d_embed
#define NO   1024  // d_mlp
#define NK   64    // components
#define NM   32    // rank m

__device__ __forceinline__ __bf16 f2bf(float f) { return (__bf16)f; }

// A fragment (16x32 bf16), ISA 7.12.2 layout, loaded from f32 row-major:
//   lane l: row = l&15 ; VGPR j<4 -> K=2j, j>=4 -> K=16+2(j-4); +8 if l>=16
template <bool RELU>
__device__ __forceinline__ v16bf load_a_frag(const float* __restrict__ src,
                                             int row_stride, int lane) {
  const int r  = lane & 15;
  const int kh = (lane >> 4) << 3;  // 0 or 8
  const float* rowp = src + (size_t)r * row_stride;
  v16bf a;
#pragma unroll
  for (int j = 0; j < 8; ++j) {
    int kb = ((j < 4) ? (2 * j) : (16 + 2 * (j - 4))) + kh;
    float v0 = rowp[kb];
    float v1 = rowp[kb + 1];
    if (RELU) { v0 = v0 > 0.f ? v0 : 0.f; v1 = v1 > 0.f ? v1 : 0.f; }
    a[2 * j]     = f2bf(v0);
    a[2 * j + 1] = f2bf(v1);
  }
  return a;
}

// B fragment (32x16 bf16) from f32 row-major: lane l: col = l&15,
// VGPR j<4 -> K=2j, j>=4 -> K=16+2(j-4); +8 if l>=16 (+1)
__device__ __forceinline__ v16bf load_b_frag(const float* __restrict__ src,
                                             int row_stride, int lane) {
  const int n   = lane & 15;
  const int kh  = (lane >> 4) << 3;  // 0 or 8
  v16bf b;
#pragma unroll
  for (int j = 0; j < 8; ++j) {
    int kb = ((j < 4) ? (2 * j) : (16 + 2 * (j - 4))) + kh;
    b[2 * j]     = f2bf(src[(size_t)(kb) * row_stride + n]);
    b[2 * j + 1] = f2bf(src[(size_t)(kb + 1) * row_stride + n]);
  }
  return b;
}

__device__ __forceinline__ v8f wmma_bf16(v16bf a, v16bf b, v8f c) {
  return __builtin_amdgcn_wmma_f32_16x16x32_bf16(false, a, false, b,
                                                 (short)0, c, false, false);
}

// ---------------------------------------------------------------------------
// pre kernel: for 4 components k..k+3 of one instance:
//   pre_ik = X_i[128 x DRED] @ W_ik[DRED x 32]
// Output stored as bf16, pre-swizzled into A-fragment layout:
//   preBF[((ik*8 + strip)*32 + lane)*16 .. +15]  (strip = batch_row/16)
// One block per 4 components; 8 waves, each wave owns a 16-row batch strip.
// Layer 2: X = h (read from d_out) with relu applied on load.
// ---------------------------------------------------------------------------
template <int DRED, bool RELU>
__global__ void pre_kernel(const float* __restrict__ X,
                           const float* __restrict__ W,
                           __bf16* __restrict__ preBF) {
  const int ikb  = blockIdx.x * 4;     // first of 4 components (i*NK + k)
  const int i    = ikb >> 6;           // /NK
  const int lane = threadIdx.x & 31;
  const int w    = threadIdx.x >> 5;   // 0..7 -> batch rows [16w, 16w+16)
  const int col  = lane & 15;
  const int rofs = (lane >> 4) << 3;

  __shared__ float lds[8][16 * 33];    // per-wave 16x32 transpose pad (+1)

  const float* xb = X + ((size_t)(16 * w) * NI + i) * DRED;  // row b=16w, d=0

  v8f c[4][2];
#pragma unroll
  for (int kk = 0; kk < 4; ++kk) { c[kk][0] = (v8f){}; c[kk][1] = (v8f){}; }

#pragma unroll 2
  for (int d0 = 0; d0 < DRED; d0 += 32) {
    v16bf a = load_a_frag<RELU>(xb + d0, NI * DRED, lane);
#pragma unroll
    for (int kk = 0; kk < 4; ++kk) {
      const float* wb = W + ((size_t)(ikb + kk) * DRED + d0) * NM;
      v16bf b0 = load_b_frag(wb,      NM, lane);
      v16bf b1 = load_b_frag(wb + 16, NM, lane);
      c[kk][0] = wmma_bf16(a, b0, c[kk][0]);
      c[kk][1] = wmma_bf16(a, b1, c[kk][1]);
    }
  }

  // C-layout -> A-layout transpose through LDS; emit bf16 fragments.
  for (int kk = 0; kk < 4; ++kk) {
    __syncthreads();
    float* T = lds[w];
#pragma unroll
    for (int j = 0; j < 8; ++j) {
      T[(j + rofs) * 33 + col]      = c[kk][0][j];
      T[(j + rofs) * 33 + 16 + col] = c[kk][1][j];
    }
    __syncthreads();
    const int r  = lane & 15;
    const int kh = (lane >> 4) << 3;
    v16bf a;
#pragma unroll
    for (int j = 0; j < 8; ++j) {
      int kb = ((j < 4) ? (2 * j) : (16 + 2 * (j - 4))) + kh;
      a[2 * j]     = f2bf(T[r * 33 + kb]);
      a[2 * j + 1] = f2bf(T[r * 33 + kb + 1]);
    }
    *(v16bf*)(preBF + (((size_t)(ikb + kk) * 8 + w) * 32 + lane) * 16) = a;
  }
}

// ---------------------------------------------------------------------------
// post kernel: for each k: post = pre_ik[128x32] @ B_ik[32xODIM] + bias,
// masked -> inner[b][i][k][o]; k-sum kept in WMMA accumulators -> sum0
// (and sum1 if non-null — `out` appears twice in the flattened return).
// Grid: (ODIM/OTILE, NB/BTILE, NI); BTILE/16 waves, each 16 rows x OTILE cols.
// A operand comes pre-swizzled in bf16: one 32B vector load per lane.
// ---------------------------------------------------------------------------
template <int ODIM, int OTILE, int BTILE>
__global__ void post_kernel(const __bf16* __restrict__ preBF,
                            const float* __restrict__ Bw,
                            const float* __restrict__ bias,
                            const unsigned char* __restrict__ mask, // numpy bool
                            float* __restrict__ inner,
                            float* __restrict__ sum0,
                            float* __restrict__ sum1) {
  constexpr int NT = OTILE / 16;
  const int i     = blockIdx.z;
  const int o0    = blockIdx.x * OTILE;
  const int b0    = blockIdx.y * BTILE;
  const int lane  = threadIdx.x & 31;
  const int w     = threadIdx.x >> 5;
  const int brow  = b0 + 16 * w;       // wave's 16-row batch strip
  const int strip = brow >> 4;         // global strip index 0..7
  const int col   = lane & 15;
  const int rofs  = (lane >> 4) << 3;

  v8f hacc[NT];
#pragma unroll
  for (int t = 0; t < NT; ++t) hacc[t] = (v8f){};

  for (int k = 0; k < NK; ++k) {
    const int ik = i * NK + k;
    v16bf a = *(const v16bf*)(preBF + (((size_t)ik * 8 + strip) * 32 + lane) * 16);

    float mv[8];
#pragma unroll
    for (int j = 0; j < 8; ++j) {
      const int b = brow + rofs + j;
      mv[j] = mask[((size_t)b * NI + i) * NK + k] ? 1.0f : 0.0f;
    }

    const float* Bk = Bw   + (size_t)ik * NM * ODIM;
    const float* bk = bias + (size_t)ik * ODIM;
#pragma unroll
    for (int t = 0; t < NT; ++t) {
      const int oc = o0 + 16 * t;
      v16bf bf = load_b_frag(Bk + oc, ODIM, lane);
      v8f c = wmma_bf16(a, bf, (v8f){});
      const float bv = bk[oc + col];
#pragma unroll
      for (int j = 0; j < 8; ++j) {
        const float v = (c[j] + bv) * mv[j];
        const int b = brow + rofs + j;
        inner[(((size_t)b * NI + i) * NK + k) * ODIM + oc + col] = v;
        hacc[t][j] += v;
      }
    }
  }

#pragma unroll
  for (int t = 0; t < NT; ++t) {
    const int oc = o0 + 16 * t + col;
#pragma unroll
    for (int j = 0; j < 8; ++j) {
      const int b = brow + rofs + j;
      const size_t idx = ((size_t)b * NI + i) * ODIM + oc;
      sum0[idx] = hacc[t][j];
      if (sum1) sum1[idx] = hacc[t][j];
    }
  }
}

// ---------------------------------------------------------------------------
// launch: x, mask, A1, B1, b1, A2, B2, b2 -> (out, [h, out], [inner1, inner2])
// workspace: preBF1 (4 MB) + preBF2 (4 MB) of d_ws.
// ---------------------------------------------------------------------------
extern "C" void kernel_launch(void* const* d_in, const int* in_sizes, int n_in,
                              void* d_out, int out_size, void* d_ws, size_t ws_size,
                              hipStream_t stream) {
  const float*         x    = (const float*)d_in[0];
  const unsigned char* mask = (const unsigned char*)d_in[1]; // bool array
  const float*         A1   = (const float*)d_in[2];
  const float*         B1   = (const float*)d_in[3];
  const float*         b1   = (const float*)d_in[4];
  const float*         A2   = (const float*)d_in[5];
  const float*         B2   = (const float*)d_in[6];
  const float*         b2   = (const float*)d_in[7];

  float* out = (float*)d_out;
  // flattened return order: out, h, out, inner1, inner2
  float* out0   = out;                                   // [128,8,256]
  float* h      = out0 + (size_t)NB * NI * ND;           // [128,8,1024]
  float* out1   = h    + (size_t)NB * NI * NO;           // [128,8,256]
  float* inner1 = out1 + (size_t)NB * NI * ND;           // [128,8,64,1024]
  float* inner2 = inner1 + (size_t)NB * NI * NK * NO;    // [128,8,64,256]

  __bf16* preBF1 = (__bf16*)d_ws;                        // [I*K][8][32][16] bf16
  __bf16* preBF2 = preBF1 + (size_t)NI * NK * NB * NM;

  // Layer 1
  pre_kernel<ND, false><<<NI * NK / 4, 256, 0, stream>>>(x, A1, preBF1);
  post_kernel<NO, 64, 128><<<dim3(NO / 64, 1, NI), 256, 0, stream>>>(
      preBF1, B1, b1, mask, inner1, h, nullptr);

  // Layer 2 (relu applied while loading h)
  pre_kernel<NO, true><<<NI * NK / 4, 256, 0, stream>>>(h, A2, preBF2);
  post_kernel<ND, 32, 64><<<dim3(ND / 32, NB / 64, NI), 128, 0, stream>>>(
      preBF2, B2, b2, mask, inner2, out0, out1);
}
